// SAModule_47090021433717
// MI455X (gfx1250) — compile-verified, matrix operand order checked by reading, not run on hardware
//
#include <hip/hip_runtime.h>
#include <math.h>
#include <stdint.h>

#define BB   8
#define PP   4096
#define MM   2048
#define KK   64
#define CIN  64
#define CC3  128
#define RR2  0.04f
#define BM   (BB * MM)
#define WAVES3 4

typedef __attribute__((ext_vector_type(16))) _Float16 v16h;
typedef __attribute__((ext_vector_type(8)))  float    v8f;

// ---------------------------------------------------------------------------
// Kernel 1: farthest point sampling (one block per batch)
// ---------------------------------------------------------------------------
__global__ __launch_bounds__(256) void fps_kernel(const float* __restrict__ pos,
                                                  float* __restrict__ pos_out,
                                                  int* __restrict__ batch_out) {
  __shared__ float px[PP], py[PP], pz[PP];
  __shared__ float rv[8];
  __shared__ int   ri[8];

  const int b    = blockIdx.x;
  const int tid  = threadIdx.x;
  const int lane = tid & 31;
  const int wid  = tid >> 5;

  for (int i = tid; i < PP; i += 256) {
    px[i] = pos[(size_t)(b * PP + i) * 3 + 0];
    py[i] = pos[(size_t)(b * PP + i) * 3 + 1];
    pz[i] = pos[(size_t)(b * PP + i) * 3 + 2];
  }
  for (int i = tid; i < MM; i += 256) batch_out[b * MM + i] = b;
  __syncthreads();

  // register-resident tile: 16 points per thread, ascending index
  float mx[16], my[16], mz[16], d[16];
  const float q0 = px[0], q1 = py[0], q2 = pz[0];
#pragma unroll
  for (int t = 0; t < 16; ++t) {
    const int p = tid * 16 + t;
    mx[t] = px[p]; my[t] = py[p]; mz[t] = pz[p];
    const float dx = mx[t] - q0, dy = my[t] - q1, dz = mz[t] - q2;
    d[t] = dx * dx + dy * dy + dz * dz;
  }
  if (tid == 0) {
    pos_out[(size_t)(b * MM) * 3 + 0] = q0;
    pos_out[(size_t)(b * MM) * 3 + 1] = q1;
    pos_out[(size_t)(b * MM) * 3 + 2] = q2;
  }

  for (int m = 1; m < MM; ++m) {
    // thread-local argmax (strict > keeps lowest index, matching jnp.argmax)
    float lv = -1.0f; int li = 0;
#pragma unroll
    for (int t = 0; t < 16; ++t)
      if (d[t] > lv) { lv = d[t]; li = tid * 16 + t; }
    // wave32 reduction
    for (int off = 16; off > 0; off >>= 1) {
      const float ov = __shfl_down(lv, off, 32);
      const int   oi = __shfl_down(li, off, 32);
      if (ov > lv || (ov == lv && oi < li)) { lv = ov; li = oi; }
    }
    if (lane == 0) { rv[wid] = lv; ri[wid] = li; }
    __syncthreads();
    float bv = rv[0]; int bi = ri[0];
#pragma unroll
    for (int w = 1; w < 8; ++w)
      if (rv[w] > bv || (rv[w] == bv && ri[w] < bi)) { bv = rv[w]; bi = ri[w]; }
    const float c0 = px[bi], c1 = py[bi], c2 = pz[bi];
    if (tid == 0) {
      pos_out[(size_t)(b * MM + m) * 3 + 0] = c0;
      pos_out[(size_t)(b * MM + m) * 3 + 1] = c1;
      pos_out[(size_t)(b * MM + m) * 3 + 2] = c2;
    }
    __syncthreads();   // protect rv/ri reuse next iteration
#pragma unroll
    for (int t = 0; t < 16; ++t) {
      const float dx = mx[t] - c0, dy = my[t] - c1, dz = mz[t] - c2;
      d[t] = fminf(d[t], dx * dx + dy * dy + dz * dz);
    }
  }
}

// ---------------------------------------------------------------------------
// Kernel 2: radius neighbor selection (one wave per center, wave32 compaction)
// ---------------------------------------------------------------------------
__global__ __launch_bounds__(256) void nbr_kernel(const float* __restrict__ pos,
                                                  const float* __restrict__ pos_out,
                                                  int* __restrict__ nbr,
                                                  unsigned long long* __restrict__ nmask) {
  const int lane   = threadIdx.x & 31;
  const int wid    = threadIdx.x >> 5;
  const int center = blockIdx.x * 8 + wid;
  const int b      = center / MM;

  const float q0 = pos_out[(size_t)center * 3 + 0];
  const float q1 = pos_out[(size_t)center * 3 + 1];
  const float q2 = pos_out[(size_t)center * 3 + 2];
  const float* pb = pos + (size_t)b * PP * 3;
  int* nrow = nbr + (size_t)center * KK;

  int cnt = 0;
  for (int c = 0; c < PP / 32 && cnt < KK; ++c) {
    const int p = c * 32 + lane;
    const float dx = pb[p * 3 + 0] - q0;
    const float dy = pb[p * 3 + 1] - q1;
    const float dz = pb[p * 3 + 2] - q2;
    const bool v = (dx * dx + dy * dy + dz * dz) <= RR2;
    const unsigned m32 = (unsigned)__ballot(v);
    if (v) {
      const int slot = cnt + __popc(m32 & ((1u << lane) - 1u));
      if (slot < KK) nrow[slot] = p;
    }
    cnt += __popc(m32);
  }
  const int filled = cnt < KK ? cnt : KK;
  for (int s = filled + lane; s < KK; s += 32) nrow[s] = 0;  // masked slots
  if (lane == 0)
    nmask[center] = (filled >= 64) ? ~0ull : ((1ull << filled) - 1ull);
}

// ---------------------------------------------------------------------------
// Kernel 3: grouped MLP via WMMA + masked max pool (one center per wave)
// ---------------------------------------------------------------------------
// Fragment loader following the CDNA5 16-bit A-matrix layout (CDNA5 ISA):
// VGPR j holds the K-pair k0 = 2j + 8*((j>>2) + hi), row/col = lane & 15.
__device__ __forceinline__ v16h load_frag(const _Float16* base, int stride,
                                          int row, int kbase, int hi) {
  union { v16h v; unsigned u[8]; } f;
  const _Float16* rp = base + row * stride + kbase;
#pragma unroll
  for (int j = 0; j < 8; ++j) {
    const int k0 = 2 * j + 8 * ((j >> 2) + hi);
    f.u[j] = *(const unsigned*)(const void*)(rp + k0);
  }
  return f.v;
}

// pack 8 f32 -> 8 f16 in one uint4 (for 16B-aligned ds_store_b128)
union Pk8 { _Float16 h[8]; uint4 u; };

__global__ __launch_bounds__(128) void mlp_kernel(const float* __restrict__ x,
                                                  const float* __restrict__ pos,
                                                  const float* __restrict__ pos_out,
                                                  const float* __restrict__ W1,
                                                  const float* __restrict__ b1,
                                                  const float* __restrict__ W2,
                                                  const float* __restrict__ b2,
                                                  const float* __restrict__ W3,
                                                  const float* __restrict__ b3,
                                                  const int* __restrict__ nbr,
                                                  const unsigned long long* __restrict__ nmask,
                                                  float* __restrict__ xout) {
  // transposed weights in LDS: Wt[n][k], so B-fragment k-pairs are 32-bit loads
  __shared__ _Float16 sW1[64 * 96];
  __shared__ _Float16 sW2[64 * 64];
  __shared__ _Float16 sW3[128 * 64];
  __shared__ float    sb1[64], sb2[64], sb3[128];
  __shared__ _Float16 hA[WAVES3][64 * 96];  // h0 (64x96), reused as h2 (64x64)
  __shared__ _Float16 hB[WAVES3][64 * 64];  // h1

  const int tid = threadIdx.x;
  for (int i = tid; i < 64 * 96; i += 128) {
    const int n = i / 96, k = i % 96;
    sW1[i] = (k < 67) ? (_Float16)W1[k * 64 + n] : (_Float16)0.f;
  }
  for (int i = tid; i < 64 * 64; i += 128) {
    const int n = i >> 6, k = i & 63;
    sW2[i] = (_Float16)W2[k * 64 + n];
  }
  for (int i = tid; i < 128 * 64; i += 128) {
    const int n = i >> 6, k = i & 63;
    sW3[i] = (_Float16)W3[k * 128 + n];
  }
  if (tid < 64)  { sb1[tid] = b1[tid]; sb2[tid] = b2[tid]; }
  if (tid < 128) { sb3[tid] = b3[tid]; }
  __syncthreads();

  const int lane = tid & 31, wid = tid >> 5;
  const int ln = lane & 15, hi = lane >> 4;
  const int center = blockIdx.x * WAVES3 + wid;
  const int b = center / MM;

  const float q0 = pos_out[(size_t)center * 3 + 0];
  const float q1 = pos_out[(size_t)center * 3 + 1];
  const float q2 = pos_out[(size_t)center * 3 + 2];
  _Float16* A  = hA[wid];
  _Float16* Bf = hB[wid];
  const int* nrow = nbr + (size_t)center * KK;

  // ---- gather h0 = [x_j | p_j - q | 0-pad] into A (64 rows x 96) ----
  // rows are 192B apart => every 8-element (16B) chunk store is aligned
  for (int n = lane; n < KK; n += 32) {
    const int j = nrow[n];
    const float* xp = x + ((size_t)b * PP + j) * CIN;
    _Float16* r = A + n * 96;
#pragma unroll
    for (int c = 0; c < 64; c += 8) {
      const float4 v0 = *(const float4*)(xp + c);
      const float4 v1 = *(const float4*)(xp + c + 4);
      Pk8 pk;
      pk.h[0] = (_Float16)v0.x; pk.h[1] = (_Float16)v0.y;
      pk.h[2] = (_Float16)v0.z; pk.h[3] = (_Float16)v0.w;
      pk.h[4] = (_Float16)v1.x; pk.h[5] = (_Float16)v1.y;
      pk.h[6] = (_Float16)v1.z; pk.h[7] = (_Float16)v1.w;
      *(uint4*)(void*)(r + c) = pk.u;
    }
    const float* pp = pos + ((size_t)b * PP + j) * 3;
    Pk8 t;
    t.h[0] = (_Float16)(pp[0] - q0);
    t.h[1] = (_Float16)(pp[1] - q1);
    t.h[2] = (_Float16)(pp[2] - q2);
    t.h[3] = (_Float16)0.f; t.h[4] = (_Float16)0.f;
    t.h[5] = (_Float16)0.f; t.h[6] = (_Float16)0.f; t.h[7] = (_Float16)0.f;
    *(uint4*)(void*)(r + 64) = t.u;
    const uint4 z = {0u, 0u, 0u, 0u};
    *(uint4*)(void*)(r + 72) = z;
    *(uint4*)(void*)(r + 80) = z;
    *(uint4*)(void*)(r + 88) = z;
  }
  __syncthreads();

  // ---- Layer 1: h1 = relu(h0 @ W1 + b1)  (64x96)@(96x64) -> Bf ----
  for (int rt = 0; rt < 4; ++rt)
    for (int ct = 0; ct < 4; ++ct) {
      v8f acc = {0.f, 0.f, 0.f, 0.f, 0.f, 0.f, 0.f, 0.f};
#pragma unroll
      for (int kt = 0; kt < 3; ++kt) {
        const v16h a = load_frag(A,   96, rt * 16 + ln, kt * 32, hi);
        const v16h w = load_frag(sW1, 96, ct * 16 + ln, kt * 32, hi);
        acc = __builtin_amdgcn_wmma_f32_16x16x32_f16(false, a, false, w,
                                                     (short)0, acc, false, false);
      }
      const int cc = ct * 16 + ln;
#pragma unroll
      for (int i = 0; i < 8; ++i) {
        const int rr = rt * 16 + 8 * hi + i;
        Bf[rr * 64 + cc] = (_Float16)fmaxf(acc[i] + sb1[cc], 0.f);
      }
    }
  __syncthreads();

  // ---- Layer 2: h2 = relu(h1 @ W2 + b2)  (64x64)@(64x64) -> A ----
  for (int rt = 0; rt < 4; ++rt)
    for (int ct = 0; ct < 4; ++ct) {
      v8f acc = {0.f, 0.f, 0.f, 0.f, 0.f, 0.f, 0.f, 0.f};
#pragma unroll
      for (int kt = 0; kt < 2; ++kt) {
        const v16h a = load_frag(Bf,  64, rt * 16 + ln, kt * 32, hi);
        const v16h w = load_frag(sW2, 64, ct * 16 + ln, kt * 32, hi);
        acc = __builtin_amdgcn_wmma_f32_16x16x32_f16(false, a, false, w,
                                                     (short)0, acc, false, false);
      }
      const int cc = ct * 16 + ln;
#pragma unroll
      for (int i = 0; i < 8; ++i) {
        const int rr = rt * 16 + 8 * hi + i;
        A[rr * 64 + cc] = (_Float16)fmaxf(acc[i] + sb2[cc], 0.f);
      }
    }
  __syncthreads();

  // ---- Layer 3 + masked max pool: out = max_k relu(h2 @ W3 + b3) ----
  const unsigned long long mv = nmask[center];
  for (int ct = 0; ct < 8; ++ct) {
    const int cc = ct * 16 + ln;
    float cmax = -INFINITY;
    for (int rt = 0; rt < 4; ++rt) {
      v8f acc = {0.f, 0.f, 0.f, 0.f, 0.f, 0.f, 0.f, 0.f};
#pragma unroll
      for (int kt = 0; kt < 2; ++kt) {
        const v16h a = load_frag(A,   64, rt * 16 + ln, kt * 32, hi);
        const v16h w = load_frag(sW3, 64, cc,           kt * 32, hi);
        acc = __builtin_amdgcn_wmma_f32_16x16x32_f16(false, a, false, w,
                                                     (short)0, acc, false, false);
      }
#pragma unroll
      for (int i = 0; i < 8; ++i) {
        const int rr = rt * 16 + 8 * hi + i;
        if ((mv >> rr) & 1ull)
          cmax = fmaxf(cmax, fmaxf(acc[i] + sb3[cc], 0.f));
      }
    }
    cmax = fmaxf(cmax, __shfl_xor(cmax, 16, 32));  // merge lane pair (same col)
    if (hi == 0) xout[(size_t)center * CC3 + cc] = cmax;
  }
}

// ---------------------------------------------------------------------------
extern "C" void kernel_launch(void* const* d_in, const int* in_sizes, int n_in,
                              void* d_out, int out_size, void* d_ws, size_t ws_size,
                              hipStream_t stream) {
  const float* x   = (const float*)d_in[0];
  const float* pos = (const float*)d_in[1];
  // d_in[2] = batch (unused; recomputed)
  const float* W1 = (const float*)d_in[3];
  const float* b1 = (const float*)d_in[4];
  const float* W2 = (const float*)d_in[5];
  const float* b2 = (const float*)d_in[6];
  const float* W3 = (const float*)d_in[7];
  const float* b3 = (const float*)d_in[8];

  float* xout     = (float*)d_out;                 // B*M x 128
  float* pos_out  = xout + (size_t)BM * CC3;       // B*M x 3
  int*   batch_out = (int*)(pos_out + (size_t)BM * 3);

  int* nbr = (int*)d_ws;                                               // B*M*K ints
  unsigned long long* nmask =
      (unsigned long long*)((char*)d_ws + (size_t)BM * KK * sizeof(int)); // B*M masks

  fps_kernel<<<BB, 256, 0, stream>>>(pos, pos_out, batch_out);
  nbr_kernel<<<BM / 8, 256, 0, stream>>>(pos, pos_out, nbr, nmask);
  mlp_kernel<<<BM / WAVES3, 128, 0, stream>>>(x, pos, pos_out,
                                              W1, b1, W2, b2, W3, b3,
                                              nbr, nmask, xout);
}
